// SemanticAttention_69956427317272
// MI455X (gfx1250) — compile-verified, matrix operand order checked by reading, not run on hardware
//
#include <hip/hip_runtime.h>

typedef __attribute__((ext_vector_type(16))) _Float16 v16h;
typedef __attribute__((ext_vector_type(8)))  float    v8f;

#define R  8
#define NN 400000
#define D  128
#define A  16

// Native CDNA5 V_TANH_F32 (TRANS op, co-executes with WMMA) when available.
#if defined(__has_builtin)
#if __has_builtin(__builtin_amdgcn_tanhf)
#define FAST_TANH(x) __builtin_amdgcn_tanhf(x)
#endif
#endif
#ifndef FAST_TANH
#define FAST_TANH(x) tanhf(x)
#endif

__global__ __launch_bounds__(256) void semattn_kernel(
    const float* __restrict__ h,
    const float* __restrict__ w1,   // [R, D, A]
    const float* __restrict__ w2,   // [R, A, 1]
    float* __restrict__ out)        // [N, D]
{
    // B fragments pre-packed in the V_WMMA_F32_16X16X32_F16 operand layout:
    // per (r, k-chunk, lane): 16 f16 = 32 contiguous bytes (2x ds_load_b128).
    __shared__ __align__(32) _Float16 bfrag[R][4][32][16];  // 32 KB
    __shared__ float w2s[R * A];                            // 512 B
    __shared__ float wlog[8][R][16];                        // 4 KB (per-wave logits/attn)

    const int tid  = threadIdx.x;
    const int wave = tid >> 5;
    const int lane = tid & 31;

    // ---- one-time per-block staging of weights (tiny, L2-resident) ----
    for (int i = 0; i < 4; ++i) {
        const int p    = tid + 256 * i;      // 1024 packets total
        const int l    = p & 31;
        const int ch   = (p >> 5) & 3;
        const int r    = p >> 7;
        const int n    = l & 15;
        const int koff = (l < 16) ? 0 : 8;
        const int kb   = ch * 32;
        #pragma unroll
        for (int e = 0; e < 16; ++e) {
            const int k = (e < 8) ? (koff + e) : (16 + koff + (e - 8));
            bfrag[r][ch][l][e] = (_Float16)w1[(r * D + (kb + k)) * A + n];
        }
    }
    if (tid < R * A) w2s[tid] = w2[tid];
    __syncthreads();

    const int n0   = (blockIdx.x * 8 + wave) * 16;  // 16 nodes per wave; 3125*8*16 == N
    const int M    = lane & 15;
    const int koff = (lane < 16) ? 0 : 8;

    // ---- phase 1: logits[r][m] via WMMA (h is streamed from HBM here, once) ----
    for (int r = 0; r < R; ++r) {
        const float* __restrict__ hrow = h + ((size_t)r * NN + (n0 + M)) * D;
        if (r < R - 1)  // warm WGP$ for the next relation's tile of this wave
            __builtin_prefetch(h + ((size_t)(r + 1) * NN + (n0 + M)) * D, 0, 0);

        v8f c = {};
        #pragma unroll
        for (int ch = 0; ch < 4; ++ch) {
            const float* pa = hrow + ch * 32 + koff;
            const float4 f0 = *(const float4*)(pa);
            const float4 f1 = *(const float4*)(pa + 4);
            const float4 f2 = *(const float4*)(pa + 16);
            const float4 f3 = *(const float4*)(pa + 20);
            v16h a;
            a[0]  = (_Float16)f0.x; a[1]  = (_Float16)f0.y; a[2]  = (_Float16)f0.z; a[3]  = (_Float16)f0.w;
            a[4]  = (_Float16)f1.x; a[5]  = (_Float16)f1.y; a[6]  = (_Float16)f1.z; a[7]  = (_Float16)f1.w;
            a[8]  = (_Float16)f2.x; a[9]  = (_Float16)f2.y; a[10] = (_Float16)f2.z; a[11] = (_Float16)f2.w;
            a[12] = (_Float16)f3.x; a[13] = (_Float16)f3.y; a[14] = (_Float16)f3.z; a[15] = (_Float16)f3.w;
            const v16h b = *(const v16h*)(&bfrag[r][ch][lane][0]);
            c = __builtin_amdgcn_wmma_f32_16x16x32_f16(
                    /*neg_a=*/false, a, /*neg_b=*/false, b,
                    /*c_mod=*/(short)0, c, /*reuse_a=*/false, /*reuse_b=*/false);
        }
        // c[j] = hidden[node j + 8*(lane>>4)][a = lane&15] (pre-activation)
        const float w2l = w2s[r * A + M];
        #pragma unroll
        for (int j = 0; j < 8; ++j) {
            float t = FAST_TANH(c[j]) * w2l;
            t += __shfl_xor(t, 1);
            t += __shfl_xor(t, 2);
            t += __shfl_xor(t, 4);
            t += __shfl_xor(t, 8);   // bits 0..3 only: stays within 16-lane half
            if (M == j) wlog[wave][r][j + ((lane >> 4) << 3)] = t;
        }
    }

    // ---- phase 2: softmax over R per node (wave-local, lanes 0..15) ----
    if (lane < 16) {
        float v[R];
        float mx = -INFINITY;
        #pragma unroll
        for (int r = 0; r < R; ++r) { v[r] = wlog[wave][r][lane]; mx = fmaxf(mx, v[r]); }
        float s = 0.f;
        #pragma unroll
        for (int r = 0; r < R; ++r) { v[r] = __expf(v[r] - mx); s += v[r]; }
        const float inv = __frcp_rn(s);
        #pragma unroll
        for (int r = 0; r < R; ++r) wlog[wave][r][lane] = v[r] * inv;
    }

    // ---- phase 3: out[n,:] = sum_r attn[r,n] * h[r,n,:] (re-read is cache-hot) ----
    for (int m = 0; m < 16; ++m) {
        float4 acc = make_float4(0.f, 0.f, 0.f, 0.f);
        #pragma unroll
        for (int r = 0; r < R; ++r) {
            const float  w  = wlog[wave][r][m];
            const float4 hv = *(const float4*)(h + ((size_t)r * NN + (n0 + m)) * D + lane * 4);
            acc.x += w * hv.x; acc.y += w * hv.y; acc.z += w * hv.z; acc.w += w * hv.w;
        }
        *(float4*)(out + (size_t)(n0 + m) * D + lane * 4) = acc;
    }
}

extern "C" void kernel_launch(void* const* d_in, const int* in_sizes, int n_in,
                              void* d_out, int out_size, void* d_ws, size_t ws_size,
                              hipStream_t stream) {
    (void)in_sizes; (void)n_in; (void)out_size; (void)d_ws; (void)ws_size;
    const float* h  = (const float*)d_in[0];
    const float* w1 = (const float*)d_in[1];
    const float* w2 = (const float*)d_in[2];
    float* out = (float*)d_out;
    dim3 grid(3125), block(256);   // 8 waves/block * 16 nodes/wave * 3125 = 400000
    hipLaunchKernelGGL(semattn_kernel, grid, block, 0, stream, h, w1, w2, out);
}
